// SummaryTransformer_1125281432104
// MI455X (gfx1250) — matrix operand order repaired and checked by reading, not compiled, not run on hardware
//
#include <hip/hip_runtime.h>
#include <cstddef>
#include <cstdint>

#define Hh   768
#define NHh  12
#define HDd  64
#define FFf  3072
#define Bb   8
#define Qq   64
#define Ss   48
#define Tt   128
#define ROWS (Bb*Qq*Ss)   /* 24576 */
#define BST  (Bb*Ss*Tt)   /* 49152 */
#define SCL  0.125f       /* 1/sqrt(64) */

typedef unsigned short u16t;
typedef __attribute__((ext_vector_type(16))) __bf16 v16bf;
typedef __attribute__((ext_vector_type(8)))  float  v8f;
typedef __attribute__((ext_vector_type(16))) u16t   v16u;
typedef __attribute__((ext_vector_type(8)))  u16t   v8u;
typedef __attribute__((ext_vector_type(4)))  unsigned int u32x4;
typedef __attribute__((ext_vector_type(8)))  int    i32x8;
typedef __attribute__((ext_vector_type(4)))  int    i32x4;

union FragU { v16u u; v16bf b; v8u h[2]; };

#if defined(__gfx1250__) && __has_builtin(__builtin_amdgcn_tensor_load_to_lds) && \
    __has_builtin(__builtin_amdgcn_s_wait_tensorcnt)
#define USE_TDM 1
#else
#define USE_TDM 0
#endif

__device__ __forceinline__ u16t f2bf(float f) {
  union { float f; unsigned u; } x; x.f = f;
  unsigned u = x.u;
  return (u16t)((u + 0x7FFFu + ((u >> 16) & 1u)) >> 16);
}

#if USE_TDM
// Issue one TDM 2D tile load: rows x row_elems bf16 elements, global row stride
// row_stride_elems, LDS destination padded by pad codes (see ISA D# group1).
// Caller must be a single wave; follow with s_wait_tensorcnt(0) before barrier.
__device__ __forceinline__ void tdm_load_2d(const u16t* gsrc, unsigned lds_byte,
                                            int rows, int row_elems,
                                            int row_stride_elems,
                                            int pad_interval_code, int pad_amount_code) {
  unsigned long long ga = (unsigned long long)(size_t)gsrc;
  u32x4 g0;
  g0.x = 1u;                                   // count=1, is_restore=0, no gather
  g0.y = lds_byte;                             // lds_addr [63:32]
  g0.z = (unsigned)ga;                         // global_addr [95:64]
  g0.w = (unsigned)((ga >> 32) & 0x1FFFFFFu)   // global_addr [120:96]
       | (2u << 30);                           // type=2 ("image") [127:126]
  unsigned td0 = (unsigned)row_stride_elems;   // tensor_dim0 (>= tile_dim0, no OOB)
  unsigned td1 = (unsigned)rows;               // tensor_dim1
  i32x8 g1;
  g1[0] = (int)((1u << 16)                           // data_size = 2 bytes
              | (1u << 20)                           // pad_enable
              | ((unsigned)pad_interval_code << 22)
              | ((unsigned)pad_amount_code << 25));
  g1[1] = (int)((td0 & 0xFFFFu) << 16);              // tensor_dim0[15:0]
  g1[2] = (int)(((td0 >> 16) & 0xFFFFu) | ((td1 & 0xFFFFu) << 16));
  g1[3] = (int)(((td1 >> 16) & 0xFFFFu) | (((unsigned)row_elems & 0xFFFFu) << 16)); // tile_dim0
  g1[4] = (int)((unsigned)rows & 0xFFFFu);           // tile_dim1 (tile_dim2 = 0)
  g1[5] = (int)td0;                                  // tensor_dim0_stride [31:0]
  g1[6] = 0;                                         // stride hi / dim1_stride lo
  g1[7] = 0;
  i32x4 z4 = {0, 0, 0, 0};
  i32x8 z8 = {0, 0, 0, 0, 0, 0, 0, 0};
  // 6-arg form (amdgpu-toolchain / clang-23): (g0, g1, g2, g3, g4, cpol)
  __builtin_amdgcn_tensor_load_to_lds(g0, g1, z4, z4, z8, 0);
}
#endif

// ---------------------------------------------------------------------------
// fp32 -> bf16 elementwise convert (weights)
// ---------------------------------------------------------------------------
__global__ void cvt_bf16_kernel(const float* __restrict__ in, u16t* __restrict__ out, int n) {
  int i = blockIdx.x * 256 + threadIdx.x;
  if (i < n) out[i] = f2bf(in[i]);
}

// ---------------------------------------------------------------------------
// Generic bf16 WMMA GEMM: C[M,N](fp32 acc) = A[M,K] x B[K,N]
//   A: fp32 (converted on the fly) or bf16 (TDM-staged), row-major.
//   B: bf16 row-major, staged transposed. 128 thr / 4 waves, 64x64 tile.
// ---------------------------------------------------------------------------
#define LDA_S 40
#define LDB_S 40

__global__ __launch_bounds__(128)
void gemm_bf16_wmma(const float* __restrict__ Af, const u16t* __restrict__ Ah,
                    const u16t* __restrict__ B, const float* __restrict__ bias,
                    float* __restrict__ outF, u16t* __restrict__ outBF,
                    int M, int N, int K, int gelu) {
  __shared__ u16t As[64 * LDA_S];
  __shared__ u16t Bt[64 * LDB_S];

  const int tid  = threadIdx.x;
  const int lane = tid & 31;
  const int wave = tid >> 5;
  const int wr   = wave >> 1, wc = wave & 1;
  const int hf   = lane >> 4;
  const int l16  = lane & 15;
  const int m0   = blockIdx.y * 64;
  const int n0   = blockIdx.x * 64;

  v8f acc[2][2] = {};

  for (int k0 = 0; k0 < K; k0 += 32) {
    // ---- stage A (64x32) ----
    if (Af) {
      #pragma unroll
      for (int j = 0; j < 4; ++j) {
        int e = tid * 16 + j * 4;
        int m = e >> 5, kk = e & 31;
        const float4 v = *(const float4*)(Af + (size_t)(m0 + m) * K + k0 + kk);
        u16t* d = &As[m * LDA_S + kk];
        d[0] = f2bf(v.x); d[1] = f2bf(v.y); d[2] = f2bf(v.z); d[3] = f2bf(v.w);
      }
    } else {
#if USE_TDM
      if (wave == 0)   // one TDM DMA moves the whole 64x32 tile with LDS padding
        tdm_load_2d(Ah + (size_t)m0 * K + k0, (unsigned)(size_t)(void*)As,
                    64, 32, K, /*interval 16 dw (64B)*/3, /*pad 4 dw (16B)*/3);
#else
      #pragma unroll
      for (int j = 0; j < 2; ++j) {
        int e = tid * 16 + j * 8;
        int m = e >> 5, kk = e & 31;
        const v8u v = *(const v8u*)(Ah + (size_t)(m0 + m) * K + k0 + kk);
        u16t* d = &As[m * LDA_S + kk];
        #pragma unroll
        for (int ii = 0; ii < 8; ++ii) d[ii] = v[ii];
      }
#endif
    }
    // ---- stage B transposed (32x64 -> Bt[n][k]) ----
    #pragma unroll
    for (int j = 0; j < 2; ++j) {
      int e = tid * 16 + j * 8;
      int kk = e >> 6, n = e & 63;
      const v8u v = *(const v8u*)(B + (size_t)(k0 + kk) * N + n0 + n);
      #pragma unroll
      for (int ii = 0; ii < 8; ++ii) Bt[(n + ii) * LDB_S + kk] = v[ii];
    }
    // prefetch next B tile into caches while we compute this step
    if (k0 + 32 < K) {
      int e = tid * 16;
      __builtin_prefetch(B + (size_t)(k0 + 32 + (e >> 6)) * N + n0 + (e & 63), 0, 1);
    }
#if USE_TDM
    if (!Af && wave == 0) __builtin_amdgcn_s_wait_tensorcnt(0);
#endif
    __syncthreads();

    #pragma unroll
    for (int i = 0; i < 2; ++i) {
      FragU a;
      const int m = wr * 32 + i * 16 + l16;
      a.h[0] = *(const v8u*)&As[m * LDA_S + hf * 8];
      a.h[1] = *(const v8u*)&As[m * LDA_S + 16 + hf * 8];
      #pragma unroll
      for (int jj = 0; jj < 2; ++jj) {
        FragU bb;
        const int n = wc * 32 + jj * 16 + l16;
        bb.h[0] = *(const v8u*)&Bt[n * LDB_S + hf * 8];
        bb.h[1] = *(const v8u*)&Bt[n * LDB_S + 16 + hf * 8];
        acc[i][jj] = __builtin_amdgcn_wmma_f32_16x16x32_bf16(
            false, a.b, false, bb.b, (short)0, acc[i][jj], false, false);
      }
    }
    __syncthreads();
  }

  #pragma unroll
  for (int i = 0; i < 2; ++i) {
    #pragma unroll
    for (int jj = 0; jj < 2; ++jj) {
      const int mb = m0 + wr * 32 + i * 16 + hf * 8;
      const int nb = n0 + wc * 32 + jj * 16 + l16;
      const float bv = bias ? bias[nb] : 0.0f;
      #pragma unroll
      for (int r = 0; r < 8; ++r) {
        float v = acc[i][jj][r] + bv;
        if (gelu) v = 0.5f * v * (1.0f + erff(v * 0.70710678118f));
        const size_t idx = (size_t)(mb + r) * N + nb;
        if (outF)  outF[idx]  = v;
        if (outBF) outBF[idx] = f2bf(v);
      }
    }
  }
}

// ---------------------------------------------------------------------------
// Attention: one block per (b,h,s). scores = qp(64x64) @ kv^T(64x128),
// scale+mask, softmax over T, res(64x64) = alpha(64x128) @ kv(128x64).
// ---------------------------------------------------------------------------
__global__ __launch_bounds__(128)
void attn_wmma_kernel(const u16t* __restrict__ qp, const u16t* __restrict__ kv,
                      const int* __restrict__ mask, u16t* __restrict__ res) {
  __shared__ u16t  qps[64 * 72];
  __shared__ u16t  kvs[128 * 72];
  __shared__ float sc[64 * 132];
  __shared__ int   ms[128];

  const int bid = blockIdx.x;
  const int s = bid % Ss;
  const int h = (bid / Ss) % NHh;
  const int b = bid / (Ss * NHh);

  const int tid  = threadIdx.x;
  const int lane = tid & 31;
  const int wave = tid >> 5;
  const int wr   = wave >> 1, wc = wave & 1;
  const int hf   = lane >> 4;
  const int l16  = lane & 15;

#if USE_TDM
  // qp tile (64x64) and kv tile (128x64): one TDM DMA each, padded rows
  // (128B data + 16B pad -> u16 stride 72).
  if (wave == 0)
    tdm_load_2d(qp + (size_t)(b * Qq) * Hh + h * HDd, (unsigned)(size_t)(void*)qps,
                64, 64, Hh, /*interval 32 dw (128B)*/4, /*pad 4 dw (16B)*/3);
  if (wave == 1)
    tdm_load_2d(kv + (size_t)(b * Ss + s) * Tt * Hh + h * HDd, (unsigned)(size_t)(void*)kvs,
                128, 64, Hh, 4, 3);
  ms[tid] = mask[(size_t)(b * Ss + s) * Tt + tid];
  if (wave < 2) __builtin_amdgcn_s_wait_tensorcnt(0);
#else
  #pragma unroll
  for (int j = 0; j < 4; ++j) {
    int e = tid * 32 + j * 8;
    int q = e >> 6, d = e & 63;
    const v8u v = *(const v8u*)(qp + (size_t)(b * Qq + q) * Hh + h * HDd + d);
    #pragma unroll
    for (int ii = 0; ii < 8; ++ii) qps[q * 72 + d + ii] = v[ii];
  }
  #pragma unroll
  for (int j = 0; j < 8; ++j) {
    int e = tid * 64 + j * 8;
    int t = e >> 6, d = e & 63;
    const v8u v = *(const v8u*)(kv + ((size_t)(b * Ss + s) * Tt + t) * Hh + h * HDd + d);
    #pragma unroll
    for (int ii = 0; ii < 8; ++ii) kvs[t * 72 + d + ii] = v[ii];
  }
  ms[tid] = mask[(size_t)(b * Ss + s) * Tt + tid];
#endif
  __syncthreads();

  // ---- GEMM1: scores = qp @ kv^T  (M=64, N=128, K=64), wave does 32x64 ----
  {
    v8f a1[2][4] = {};
    #pragma unroll
    for (int k0 = 0; k0 < 64; k0 += 32) {
      #pragma unroll
      for (int i = 0; i < 2; ++i) {
        FragU a;
        const int m = wr * 32 + i * 16 + l16;
        a.h[0] = *(const v8u*)&qps[m * 72 + k0 + hf * 8];
        a.h[1] = *(const v8u*)&qps[m * 72 + k0 + 16 + hf * 8];
        #pragma unroll
        for (int jj = 0; jj < 4; ++jj) {
          FragU bb;
          const int n = wc * 64 + jj * 16 + l16;   // t index; kvs already "B transposed"
          bb.h[0] = *(const v8u*)&kvs[n * 72 + k0 + hf * 8];
          bb.h[1] = *(const v8u*)&kvs[n * 72 + k0 + 16 + hf * 8];
          a1[i][jj] = __builtin_amdgcn_wmma_f32_16x16x32_bf16(
              false, a.b, false, bb.b, (short)0, a1[i][jj], false, false);
        }
      }
    }
    #pragma unroll
    for (int i = 0; i < 2; ++i)
      #pragma unroll
      for (int jj = 0; jj < 4; ++jj) {
        const int mb = wr * 32 + i * 16 + hf * 8;
        const int nb = wc * 64 + jj * 16 + l16;
        const float mk = (float)ms[nb] * -10000.0f;
        #pragma unroll
        for (int r = 0; r < 8; ++r)
          sc[(mb + r) * 132 + nb] = a1[i][jj][r] * SCL + mk;
      }
  }
  __syncthreads();

  // ---- softmax over T=128, pack alpha bf16 in-place over sc rows ----
  if (tid < 64) {
    const int r = tid;
    float mx = -3.4e38f;
    for (int t = 0; t < 128; ++t) mx = fmaxf(mx, sc[r * 132 + t]);
    float sum = 0.0f;
    for (int t = 0; t < 128; ++t) {
      float e = expf(sc[r * 132 + t] - mx);
      sc[r * 132 + t] = e; sum += e;
    }
    const float inv = 1.0f / sum;
    u16t* ar = (u16t*)&sc[r * 132];
    for (int t = 0; t < 128; t += 2) {
      float a0 = sc[r * 132 + t] * inv;
      float a1v = sc[r * 132 + t + 1] * inv;
      ar[t] = f2bf(a0); ar[t + 1] = f2bf(a1v);
    }
  }
  __syncthreads();

  // ---- GEMM2: res = alpha(64x128) @ kv(128x64), wave does 32x32 ----
  {
    const u16t* al = (const u16t*)sc;   // row stride 264 u16
    v8f a2[2][2] = {};
    #pragma unroll
    for (int k0 = 0; k0 < 128; k0 += 32) {
      #pragma unroll
      for (int i = 0; i < 2; ++i) {
        FragU a;
        const int m = wr * 32 + i * 16 + l16;
        a.h[0] = *(const v8u*)&al[m * 264 + k0 + hf * 16];
        a.h[1] = *(const v8u*)&al[m * 264 + k0 + hf * 16 + 8];
        #pragma unroll
        for (int jj = 0; jj < 2; ++jj) {
          FragU bb;
          const int n = wc * 32 + jj * 16 + l16;   // d index; kvs is [t][d]
          #pragma unroll
          for (int e2 = 0; e2 < 16; ++e2)
            bb.u[e2] = kvs[(k0 + hf * 16 + e2) * 72 + n];
          a2[i][jj] = __builtin_amdgcn_wmma_f32_16x16x32_bf16(
              false, a.b, false, bb.b, (short)0, a2[i][jj], false, false);
        }
      }
    }
    #pragma unroll
    for (int i = 0; i < 2; ++i)
      #pragma unroll
      for (int jj = 0; jj < 2; ++jj) {
        const int mb = wr * 32 + i * 16 + hf * 8;
        const int nb = wc * 32 + jj * 16 + l16;
        #pragma unroll
        for (int r = 0; r < 8; ++r) {
          const int mq = mb + r;
          res[((size_t)(b * Qq + mq) * Ss + s) * Hh + h * HDd + nb] = f2bf(a2[i][jj][r]);
        }
      }
  }
}

// ---------------------------------------------------------------------------
// LayerNorm over 768 cols; optional residual in2; fp32 and/or bf16 output.
// ---------------------------------------------------------------------------
__global__ __launch_bounds__(256)
void ln_kernel(const float* __restrict__ in1, const float* __restrict__ in2,
               const float* __restrict__ g, const float* __restrict__ bta,
               float* __restrict__ outF, u16t* __restrict__ outBF) {
  __shared__ float rs[256], rq[256];
  const int row = blockIdx.x;
  const int tid = threadIdx.x;
  float x[3], sum = 0.0f, sq = 0.0f;
  #pragma unroll
  for (int j = 0; j < 3; ++j) {
    int c = tid + j * 256;
    float v = in1[(size_t)row * Hh + c];
    if (in2) v += in2[(size_t)row * Hh + c];
    x[j] = v; sum += v; sq += v * v;
  }
  rs[tid] = sum; rq[tid] = sq; __syncthreads();
  for (int st = 128; st > 0; st >>= 1) {
    if (tid < st) { rs[tid] += rs[tid + st]; rq[tid] += rq[tid + st]; }
    __syncthreads();
  }
  const float mean = rs[0] * (1.0f / 768.0f);
  const float var  = fmaxf(rq[0] * (1.0f / 768.0f) - mean * mean, 0.0f);
  const float rinv = rsqrtf(var + 1e-12f);
  #pragma unroll
  for (int j = 0; j < 3; ++j) {
    int c = tid + j * 256;
    float y = (x[j] - mean) * rinv * g[c] + bta[c];
    if (outF)  outF[(size_t)row * Hh + c]  = y;
    if (outBF) outBF[(size_t)row * Hh + c] = f2bf(y);
  }
}

// ---------------------------------------------------------------------------
extern "C" void kernel_launch(void* const* d_in, const int* in_sizes, int n_in,
                              void* d_out, int out_size, void* d_ws, size_t ws_size,
                              hipStream_t stream) {
  const float* q    = (const float*)d_in[0];
  const float* k    = (const float*)d_in[1];
  const int*   mask = (const int*)d_in[2];
  const float* Wq   = (const float*)d_in[3];
  const float* Wkv  = (const float*)d_in[4];
  const float* Wo   = (const float*)d_in[5];
  const float* bo   = (const float*)d_in[6];
  const float* ln1g = (const float*)d_in[7];
  const float* ln1b = (const float*)d_in[8];
  const float* W1   = (const float*)d_in[9];
  const float* b1   = (const float*)d_in[10];
  const float* W2   = (const float*)d_in[11];
  const float* b2   = (const float*)d_in[12];
  const float* ln2g = (const float*)d_in[13];
  const float* ln2b = (const float*)d_in[14];
  (void)in_sizes; (void)n_in; (void)out_size; (void)ws_size;

  char* ws = (char*)d_ws;
  size_t off = 0;
  auto alloc = [&](size_t bytes) -> void* {
    void* p = ws + off;
    off = (off + bytes + 255) & ~(size_t)255;
    return p;
  };

  u16t*  wq_bf   = (u16t*)alloc((size_t)Hh * Hh * 2);
  u16t*  wkv_bf  = (u16t*)alloc((size_t)Hh * Hh * 2);
  u16t*  wo_bf   = (u16t*)alloc((size_t)Hh * Hh * 2);
  u16t*  w1_bf   = (u16t*)alloc((size_t)Hh * FFf * 2);
  u16t*  w2_bf   = (u16t*)alloc((size_t)FFf * Hh * 2);
  u16t*  qp_bf   = (u16t*)alloc((size_t)Bb * Qq * Hh * 2);
  u16t*  kv_bf   = (u16t*)alloc((size_t)BST * Hh * 2);
  u16t*  res_bf  = (u16t*)alloc((size_t)ROWS * Hh * 2);
  float* attnpre = (float*)alloc((size_t)ROWS * Hh * 4);
  float* attn_f  = (float*)alloc((size_t)ROWS * Hh * 4);
  u16t*  attn_bf = (u16t*)alloc((size_t)ROWS * Hh * 2);
  u16t*  h1_bf   = (u16t*)alloc((size_t)ROWS * FFf * 2);
  float* ffn_f   = (float*)alloc((size_t)ROWS * Hh * 4);

  // 1) weights -> bf16
  cvt_bf16_kernel<<<(Hh * Hh + 255) / 256, 256, 0, stream>>>(Wq,  wq_bf,  Hh * Hh);
  cvt_bf16_kernel<<<(Hh * Hh + 255) / 256, 256, 0, stream>>>(Wkv, wkv_bf, Hh * Hh);
  cvt_bf16_kernel<<<(Hh * Hh + 255) / 256, 256, 0, stream>>>(Wo,  wo_bf,  Hh * Hh);
  cvt_bf16_kernel<<<(Hh * FFf + 255) / 256, 256, 0, stream>>>(W1, w1_bf, Hh * FFf);
  cvt_bf16_kernel<<<(FFf * Hh + 255) / 256, 256, 0, stream>>>(W2, w2_bf, FFf * Hh);

  // 2) qp = q @ Wq          [512,768]
  gemm_bf16_wmma<<<dim3(Hh / 64, (Bb * Qq) / 64), 128, 0, stream>>>(
      q, nullptr, wq_bf, nullptr, nullptr, qp_bf, Bb * Qq, Hh, Hh, 0);
  // 3) kv = k @ Wkv         [49152,768]
  gemm_bf16_wmma<<<dim3(Hh / 64, BST / 64), 128, 0, stream>>>(
      k, nullptr, wkv_bf, nullptr, nullptr, kv_bf, BST, Hh, Hh, 0);
  // 4) attention            -> res_bf [24576,768]
  attn_wmma_kernel<<<Bb * NHh * Ss, 128, 0, stream>>>(qp_bf, kv_bf, mask, res_bf);
  // 5) res @ Wo + bo        -> attnpre fp32
  gemm_bf16_wmma<<<dim3(Hh / 64, ROWS / 64), 128, 0, stream>>>(
      nullptr, res_bf, wo_bf, bo, attnpre, nullptr, ROWS, Hh, Hh, 0);
  // 6) LN1                  -> attn_f fp32 + attn_bf bf16
  ln_kernel<<<ROWS, 256, 0, stream>>>(attnpre, nullptr, ln1g, ln1b, attn_f, attn_bf);
  // 7) gelu(attn @ W1 + b1) -> h1_bf [24576,3072]
  gemm_bf16_wmma<<<dim3(FFf / 64, ROWS / 64), 128, 0, stream>>>(
      nullptr, attn_bf, w1_bf, b1, nullptr, h1_bf, ROWS, FFf, Hh, 1);
  // 8) h1 @ W2 + b2         -> ffn_f fp32
  gemm_bf16_wmma<<<dim3(Hh / 64, ROWS / 64), 128, 0, stream>>>(
      nullptr, h1_bf, w2_bf, b2, ffn_f, nullptr, ROWS, Hh, FFf, 0);
  // 9) LN2(attn + ffn)      -> d_out fp32
  ln_kernel<<<ROWS, 256, 0, stream>>>(attn_f, ffn_f, ln2g, ln2b, (float*)d_out, nullptr);
}
